// ScriptAwareAttention_41764261986740
// MI455X (gfx1250) — compile-verified, matrix operand order checked by reading, not run on hardware
//
#include <hip/hip_runtime.h>
#include <math.h>

// Problem constants (match reference)
constexpr int BATCH = 2;
constexpr int SEQ   = 2048;
constexpr int HID   = 1024;
constexpr int NHEAD = 16;
constexpr int HDIM  = 64;   // HID / NHEAD

typedef __attribute__((ext_vector_type(16))) __bf16 v16bf;
typedef __attribute__((ext_vector_type(8)))  float  v8f;
typedef __attribute__((ext_vector_type(4)))  unsigned int su4;
typedef __attribute__((ext_vector_type(8)))  unsigned int su8;

union ABfrag { v16bf v; unsigned int u[8]; };
union CDfrag { v8f  v; float f[8]; };

__device__ __forceinline__ unsigned short f2bf(float f) {
    union { float f; unsigned int u; } x; x.f = f;
    unsigned int u = x.u;
    unsigned int r = u + 0x7FFFu + ((u >> 16) & 1u);   // round-to-nearest-even
    return (unsigned short)(r >> 16);
}

// A-fragment K offset per ISA 16-bit A 16x32 layout:
// lanes 0-15: VGPR0..3 -> K=0..7, VGPR4..7 -> K=16..23 ; lanes 16-31: +8
__device__ __forceinline__ int kofs(int v, int half) {
    return (v < 4) ? (8 * half + 2 * v) : (16 + 8 * half + 2 * (v - 4));
}

// ---------------------------------------------------------------------------
// GEMM with bias:  Out[m,n] = sum_k A[m,k] * W[n,k] + bias[n]
// BM=128, BN=128, BK=64. 256 threads = 8 waves; wave (wr,wc) owns 32x64:
// 2 A-frags x 4 B-frags -> 16 WMMAs per K-step (B-frags reused twice).
// ---------------------------------------------------------------------------
constexpr int BM = 128, BN = 128, BK = 64;

template <bool A_F32, bool OUT_F32>
__global__ __launch_bounds__(256)
void gemm_bias(const void* __restrict__ Aptr, const float* __restrict__ W,
               const float* __restrict__ bias, void* __restrict__ Out,
               int M, int N, int K)
{
    __shared__ unsigned short Al[BM * 66];   // [row][k], stride 66 halfwords
    __shared__ unsigned short Bl[BN * 66];   // [n][k],   stride 66 halfwords

    const int tid  = threadIdx.x;
    const int lane = tid & 31;
    const int wave = tid >> 5;
    const int wr   = wave & 3;     // row quarter (32 rows)
    const int wc   = wave >> 2;    // col half   (64 cols)
    const int half = lane >> 4;
    const int l16  = lane & 15;

    const int mbase = blockIdx.x * BM;
    const int nbase = blockIdx.y * BN;

    CDfrag acc[8];
#pragma unroll
    for (int t = 0; t < 8; ++t)
#pragma unroll
        for (int i = 0; i < 8; ++i) acc[t].f[i] = 0.f;

    const int srow = tid >> 1;                 // 0..127
    const int scol = (tid & 1) * 32;           // 0 or 32

    for (int kb = 0; kb < K; kb += BK) {
        // ---- stage A tile (convert f32 -> bf16 if needed) ----
        if (A_F32) {
            const float* Af = (const float*)Aptr;
#pragma unroll
            for (int j = 0; j < 32; ++j)
                Al[srow * 66 + scol + j] =
                    f2bf(Af[(size_t)(mbase + srow) * K + kb + scol + j]);
        } else {
            const unsigned short* Ab = (const unsigned short*)Aptr;
#pragma unroll
            for (int j = 0; j < 32; ++j)
                Al[srow * 66 + scol + j] =
                    Ab[(size_t)(mbase + srow) * K + kb + scol + j];
        }
        // ---- stage W tile: Bl[n][k] ----
#pragma unroll
        for (int j = 0; j < 32; ++j)
            Bl[srow * 66 + scol + j] =
                f2bf(W[(size_t)(nbase + srow) * K + kb + scol + j]);

        __syncthreads();

#pragma unroll
        for (int c = 0; c < 2; ++c) {
            ABfrag a0, a1;
            const int row0 = wr * 32 + l16;
#pragma unroll
            for (int v = 0; v < 8; ++v) {
                const int k0 = c * 32 + kofs(v, half);
                a0.u[v] = *(const unsigned int*)&Al[row0 * 66 + k0];
                a1.u[v] = *(const unsigned int*)&Al[(row0 + 16) * 66 + k0];
            }
#pragma unroll
            for (int ct = 0; ct < 4; ++ct) {
                ABfrag bfr;
                const int nrow = wc * 64 + ct * 16 + l16;
#pragma unroll
                for (int v = 0; v < 8; ++v)
                    bfr.u[v] = *(const unsigned int*)
                        &Bl[nrow * 66 + c * 32 + kofs(v, half)];
                acc[ct].v = __builtin_amdgcn_wmma_f32_16x16x32_bf16(
                    false, a0.v, false, bfr.v, (short)0, acc[ct].v, false, false);
                acc[4 + ct].v = __builtin_amdgcn_wmma_f32_16x16x32_bf16(
                    false, a1.v, false, bfr.v, (short)0, acc[4 + ct].v, false, false);
            }
        }
        __syncthreads();
    }

    // ---- epilogue: bias + store (C layout: lane = col, VGPR r = row) ----
#pragma unroll
    for (int s = 0; s < 2; ++s)
#pragma unroll
        for (int ct = 0; ct < 4; ++ct) {
            const int ncol = nbase + wc * 64 + ct * 16 + l16;
            const float bv = bias[ncol];
#pragma unroll
            for (int r = 0; r < 8; ++r) {
                const int mrow = mbase + wr * 32 + s * 16 + r + 8 * half;
                const float val = acc[s * 4 + ct].f[r] + bv;
                if (OUT_F32)
                    ((float*)Out)[(size_t)mrow * N + ncol] = val;
                else
                    ((unsigned short*)Out)[(size_t)mrow * N + ncol] = f2bf(val);
            }
        }
}

// ---------------------------------------------------------------------------
// Flash attention over projected Q/K/V (bf16, [b, s, HID], head h at columns
// h*HDIM..+63). Grid: (SEQ/64, NHEAD, BATCH), 128 threads (4 waves).
// K tile is staged by the Tensor Data Mover (tensor_load_to_lds, TENSORcnt);
// V tile is manually transposed into LDS. Each wave owns 16 query rows.
// ---------------------------------------------------------------------------
constexpr int KT = 64;   // key tile

__global__ __launch_bounds__(128)
void flash_attn(const unsigned short* __restrict__ Qp,
                const unsigned short* __restrict__ Kp,
                const unsigned short* __restrict__ Vp,
                const int*   __restrict__ mask,
                const float* __restrict__ smask,
                unsigned short* __restrict__ ctx)
{
    __shared__ unsigned short Kl[KT * 66];        // [key][d], stride 66 halves
                                                  //  = 128B row + 4B pad (TDM pad)
    __shared__ unsigned short Vl[HDIM * 66];      // [d][key] (pairs along key)
    __shared__ float          Sl[4][16 * 66];     // per-wave scores [row][col]
    __shared__ unsigned short Pl[4][16 * 66];     // per-wave probs  [row][col]
    __shared__ float Mrow[4][16], Lrow[4][16], Crow[4][16];

    const int tid  = threadIdx.x;
    const int lane = tid & 31;
    const int wave = tid >> 5;
    const int half = lane >> 4;
    const int l16  = lane & 15;

    const int b = blockIdx.z;
    const int h = blockIdx.y;
    const int qbase = blockIdx.x * 64 + wave * 16;

    // ---- Q fragments (A layout), direct global gather, loaded once ----
    ABfrag aq[2];
    const unsigned short* Qrow =
        Qp + (size_t)(b * SEQ + qbase + l16) * HID + h * HDIM;
#pragma unroll
    for (int c = 0; c < 2; ++c)
#pragma unroll
        for (int v = 0; v < 8; ++v)
            aq[c].u[v] = *(const unsigned int*)&Qrow[c * 32 + kofs(v, half)];

    CDfrag o[4];
#pragma unroll
    for (int t = 0; t < 4; ++t)
#pragma unroll
        for (int i = 0; i < 8; ++i) o[t].f[i] = 0.f;

    if (lane < 16) {
        Mrow[wave][lane] = -__builtin_inff();
        Lrow[wave][lane] = 0.f;
    }

    for (int kb = 0; kb < SEQ; kb += KT) {
        __syncthreads();   // previous-iteration consumers done with Kl/Vl

        // ---- K tile via Tensor Data Mover: 64x64 bf16 2D tile, row stride
        //      HID elements, LDS pad 1 dword per 32 dwords (-> stride 66) ----
        if (wave == 0) {
            const unsigned long long ga = (unsigned long long)
                (const void*)(Kp + (size_t)(b * SEQ + kb) * HID + h * HDIM);
            const unsigned kloff =
                (unsigned)(unsigned long long)(const void*)&Kl[0];
            su4 g0;
            g0[0] = 1u;                                       // count=1
            g0[1] = kloff;                                    // lds_addr
            g0[2] = (unsigned)(ga & 0xFFFFFFFFu);             // global_addr lo
            g0[3] = (unsigned)((ga >> 32) & 0x1FFFFFFu)       // global_addr hi
                  | (2u << 30);                               // type=2 (image)
            su8 g1;
            g1[0] = (1u << 16)            // data_size = 2 bytes
                  | (1u << 20)            // pad_enable
                  | (4u << 22);           // pad_interval: 32 dwords
                                          // pad_amount encoded 0 = 1 dword
            g1[1] = ((unsigned)HID & 0xFFFFu) << 16;          // tensor_dim0 lo16
            g1[2] = ((unsigned)HID >> 16)                     // tensor_dim0 hi16
                  | (((unsigned)(BATCH * SEQ) & 0xFFFFu) << 16); // tensor_dim1 lo
            g1[3] = ((unsigned)(BATCH * SEQ) >> 16)           // tensor_dim1 hi16
                  | (64u << 16);                              // tile_dim0 = 64
            g1[4] = 64u;                                      // tile_dim1 = 64
            g1[5] = (unsigned)HID;                            // dim0_stride lo32
            g1[6] = 0u;                                       // stride hi / dim1_stride
            g1[7] = 0u;
            asm volatile("tensor_load_to_lds %0, %1"
                         :: "s"(g0), "s"(g1) : "memory");
            __builtin_amdgcn_s_wait_tensorcnt(0);
        }

        // ---- stage V tile transposed: Vl[d][key] ----
        for (int t = tid; t < KT * (HDIM / 2); t += 128) {
            const int key = t >> 5, dw = t & 31;
            const unsigned int vv = *(const unsigned int*)
                &Vp[(size_t)(b * SEQ + kb + key) * HID + h * HDIM + 2 * dw];
            Vl[(2 * dw + 0) * 66 + key] = (unsigned short)(vv & 0xFFFFu);
            Vl[(2 * dw + 1) * 66 + key] = (unsigned short)(vv >> 16);
        }
        // prefetch next tile (global_prefetch_b8)
        if (kb + KT < SEQ && tid < 64) {
            const size_t nb = (size_t)(b * SEQ + kb + KT + tid) * HID + h * HDIM;
            __builtin_prefetch(&Kp[nb], 0, 1);
            __builtin_prefetch(&Vp[nb], 0, 1);
        }
        __syncthreads();

        // ---- scores: S = Q * K^T  (B[d][n] = K[kb+ct*16+n][d]) ----
        CDfrag sc[4];
#pragma unroll
        for (int t = 0; t < 4; ++t)
#pragma unroll
            for (int i = 0; i < 8; ++i) sc[t].f[i] = 0.f;

#pragma unroll
        for (int ct = 0; ct < 4; ++ct)
#pragma unroll
            for (int c = 0; c < 2; ++c) {
                ABfrag bk;
#pragma unroll
                for (int v = 0; v < 8; ++v)
                    bk.u[v] = *(const unsigned int*)
                        &Kl[(ct * 16 + l16) * 66 + c * 32 + kofs(v, half)];
                sc[ct].v = __builtin_amdgcn_wmma_f32_16x16x32_bf16(
                    false, aq[c].v, false, bk.v, (short)0, sc[ct].v, false, false);
            }

        // ---- scale, script mask (mult), mask==0 -> -inf; dump to LDS ----
#pragma unroll
        for (int ct = 0; ct < 4; ++ct) {
            const int col  = ct * 16 + l16;
            const int kidx = kb + col;
#pragma unroll
            for (int r = 0; r < 8; ++r) {
                const int row = r + 8 * half;
                const size_t off = ((size_t)b * SEQ + qbase + row) * SEQ + kidx;
                float s = sc[ct].f[r] * 0.125f * smask[off];   // 1/sqrt(64)
                if (mask[off] == 0) s = -__builtin_inff();
                Sl[wave][row * 66 + col] = s;
            }
        }
        __syncthreads();

        // ---- online softmax row scan: lanes 0..15 own one row each ----
        if (lane < 16) {
            const int row = lane;
            const float m_old = Mrow[wave][row];
            float mx = m_old;
            for (int c2 = 0; c2 < KT; ++c2)
                mx = fmaxf(mx, Sl[wave][row * 66 + c2]);
            const bool finite = (mx > -__builtin_inff());
            const float corr =
                (m_old > -__builtin_inff()) ? __expf(m_old - mx) : 0.f;
            float lsum = 0.f;
            for (int c2 = 0; c2 < KT; ++c2) {
                const float s = Sl[wave][row * 66 + c2];
                const float p = finite ? __expf(s - mx) : 0.f;
                lsum += p;
                Pl[wave][row * 66 + c2] = f2bf(p);
            }
            Mrow[wave][row] = mx;
            Lrow[wave][row] = Lrow[wave][row] * corr + lsum;
            Crow[wave][row] = corr;
        }
        __syncthreads();

        // ---- rescale O by corr, then O += P * V ----
        float cr[8];
#pragma unroll
        for (int r = 0; r < 8; ++r) cr[r] = Crow[wave][r + 8 * half];
#pragma unroll
        for (int ct = 0; ct < 4; ++ct)
#pragma unroll
            for (int r = 0; r < 8; ++r) o[ct].f[r] *= cr[r];

#pragma unroll
        for (int c = 0; c < 2; ++c) {
            ABfrag ap;   // A = P (16 x 64), row = l16
#pragma unroll
            for (int v = 0; v < 8; ++v)
                ap.u[v] = *(const unsigned int*)
                    &Pl[wave][l16 * 66 + c * 32 + kofs(v, half)];
#pragma unroll
            for (int ct = 0; ct < 4; ++ct) {
                ABfrag bv;   // B = V, B[kk][n] = Vl[d][kk]
#pragma unroll
                for (int v = 0; v < 8; ++v)
                    bv.u[v] = *(const unsigned int*)
                        &Vl[(ct * 16 + l16) * 66 + c * 32 + kofs(v, half)];
                o[ct].v = __builtin_amdgcn_wmma_f32_16x16x32_bf16(
                    false, ap.v, false, bv.v, (short)0, o[ct].v, false, false);
            }
        }
    }
    __syncthreads();

    // ---- normalize by row sums and store context (bf16, [b,s,HID]) ----
    float inv_l[8];
#pragma unroll
    for (int r = 0; r < 8; ++r) {
        const float L = Lrow[wave][r + 8 * half];
        inv_l[r] = (L > 0.f) ? (1.f / L) : 0.f;
    }
#pragma unroll
    for (int ct = 0; ct < 4; ++ct) {
        const int col = h * HDIM + ct * 16 + l16;
#pragma unroll
        for (int r = 0; r < 8; ++r) {
            const int q = qbase + r + 8 * half;
            ctx[(size_t)(b * SEQ + q) * HID + col] = f2bf(o[ct].f[r] * inv_l[r]);
        }
    }
}

// ---------------------------------------------------------------------------
extern "C" void kernel_launch(void* const* d_in, const int* in_sizes, int n_in,
                              void* d_out, int out_size, void* d_ws, size_t ws_size,
                              hipStream_t stream)
{
    const float* query = (const float*)d_in[0];
    const float* key_  = (const float*)d_in[1];
    const float* value = (const float*)d_in[2];
    const int*   mask  = (const int*)d_in[3];
    const float* smask = (const float*)d_in[4];
    const float* Wq = (const float*)d_in[5];
    const float* bq = (const float*)d_in[6];
    const float* Wk = (const float*)d_in[7];
    const float* bk = (const float*)d_in[8];
    const float* Wv = (const float*)d_in[9];
    const float* bv = (const float*)d_in[10];
    const float* Wo = (const float*)d_in[11];
    const float* bo = (const float*)d_in[12];
    float* out = (float*)d_out;

    const int M = BATCH * SEQ;                     // 4096
    const size_t mat = (size_t)M * HID;            // 4M elements

    unsigned short* qp = (unsigned short*)d_ws;    // bf16 buffers in workspace
    unsigned short* kp = qp + mat;
    unsigned short* vp = kp + mat;
    unsigned short* cx = vp + mat;                 // total 32 MB of ws

    dim3 gg(M / BM, HID / BN);                     // 32 x 8
    dim3 bb(256);
    gemm_bias<true, false><<<gg, bb, 0, stream>>>(query, Wq, bq, qp, M, HID, HID);
    gemm_bias<true, false><<<gg, bb, 0, stream>>>(key_,  Wk, bk, kp, M, HID, HID);
    gemm_bias<true, false><<<gg, bb, 0, stream>>>(value, Wv, bv, vp, M, HID, HID);

    dim3 ga(SEQ / 64, NHEAD, BATCH);               // 32 x 16 x 2
    flash_attn<<<ga, dim3(128), 0, stream>>>(qp, kp, vp, mask, smask, cx);

    gemm_bias<false, true><<<gg, bb, 0, stream>>>(cx, Wo, bo, out, M, HID, HID);
}
